// DeformableDecoderRPN_11407433138644
// MI455X (gfx1250) — matrix operand order repaired and checked by reading, not compiled, not run on hardware
//
#include <hip/hip_runtime.h>

typedef __bf16 bf16;
typedef __attribute__((ext_vector_type(8)))  bf16  v8bf;
typedef __attribute__((ext_vector_type(16))) bf16  v16bf;
typedef __attribute__((ext_vector_type(8)))  float v8f;

#define BNS 0.99950037f  /* 1/sqrt(1+1e-3) */

union bfrag { v16bf v; v8bf h[2]; };

// ---------------------------------------------------------------------------
// Implicit-GEMM 3x3 conv, bf16 NHWC (halo-padded, pitch W+2), f32 accum WMMA.
// Weights packed [Cout][9*Cin], k = tap*Cin + cin.
// Block = 128 threads = 4 waves laid out 2(M) x 2(N); each wave owns a
// 32 x (NJ*16) output tile (2 x NJ WMMA register tile).
//   NJ=4: block covers 64(M) x 128(N)  -- main layers (Cout % 128 == 0)
//   NJ=2: block covers 64(M) x  64(N)  -- heatmap conv (Cout == 64)
// Input halo is pre-zeroed: no bounds checks, EXEC stays full, inner loop is
// pure clause'd b128 loads + 2*NJ v_wmma per K-chunk.
// ---------------------------------------------------------------------------
template <int NJ>
__global__ __launch_bounds__(128)
void conv3x3_wmma(const bf16* __restrict__ in, const bf16* __restrict__ wpk,
                  const float* __restrict__ bias, bf16* __restrict__ out,
                  int Win, int Wo, int Cin, int Cout, int stride,
                  float scale, int relu)
{
    const int lane = threadIdx.x & 31;
    const int wave = threadIdx.x >> 5;
    const int hf   = lane >> 4;    // lane-half: K-phase selector
    const int l16  = lane & 15;
    const int Wp   = Win + 2;      // padded input pitch
    const int Wop  = Wo + 2;       // padded output pitch
    const int oy   = blockIdx.y;
    const int m0   = blockIdx.x * 64 + (wave & 1) * 32;
    const int n0   = blockIdx.z * (32 * NJ) + (wave >> 1) * (16 * NJ);
    const int Ktot = 9 * Cin;

    v8f acc[2][NJ] = {};

    const int ox0 = (m0 + l16) * stride;
    const int ox1 = (m0 + 16 + l16) * stride;
    const bf16* wb[NJ];
#pragma unroll
    for (int j = 0; j < NJ; ++j)
        wb[j] = wpk + (size_t)(n0 + j * 16 + l16) * Ktot + 16 * hf;

    for (int tap = 0; tap < 9; ++tap) {
        const int dy = tap / 3, dx = tap % 3;       // padded-space offsets
        const size_t rowb = (size_t)(oy * stride + dy) * Wp;
        const bf16* a0 = in + (rowb + ox0 + dx) * Cin + 16 * hf;
        const bf16* a1 = in + (rowb + ox1 + dx) * Cin + 16 * hf;

        for (int c0 = 0; c0 < Cin; c0 += 32) {
            bfrag A[2], B[NJ];
            // A 16x32 fragment: per-lane K run 16*hf..16*hf+15, contiguous.
            A[0].h[0] = *(const v8bf*)(a0 + c0);
            A[0].h[1] = *(const v8bf*)(a0 + c0 + 8);
            A[1].h[0] = *(const v8bf*)(a1 + c0);
            A[1].h[1] = *(const v8bf*)(a1 + c0 + 8);
#pragma unroll
            for (int j = 0; j < NJ; ++j) {
                // B 32x16 fragment: per-lane K run 16*hf..16*hf+15, contiguous.
                const bf16* bj = wb[j] + tap * Cin + c0;
                B[j].h[0] = *(const v8bf*)(bj);
                B[j].h[1] = *(const v8bf*)(bj + 8);
            }
            __builtin_prefetch(wb[0] + tap * Cin + c0 + 32, 0, 1);
#pragma unroll
            for (int i = 0; i < 2; ++i)
#pragma unroll
                for (int j = 0; j < NJ; ++j)
                    acc[i][j] = __builtin_amdgcn_wmma_f32_16x16x32_bf16(
                        false, A[i].v, false, B[j].v, (short)0, acc[i][j],
                        false, false);
        }
    }

    // D 16x16: lane l16 -> col n; VGPR r -> row m = r + 8*hf
#pragma unroll
    for (int j = 0; j < NJ; ++j) {
        const int n = n0 + j * 16 + l16;
        const float bb = bias ? bias[n] : 0.0f;
#pragma unroll
        for (int i = 0; i < 2; ++i) {
#pragma unroll
            for (int r = 0; r < 8; ++r) {
                const int mx = m0 + i * 16 + r + 8 * hf;
                float v = (acc[i][j][r] + bb) * scale;
                if (relu) v = fmaxf(v, 0.0f);
                out[((size_t)(oy + 1) * Wop + mx + 1) * Cout + n] = (bf16)v;
            }
        }
    }
}

// --- zero fill (for halo-padded buffers) ------------------------------------
__global__ void zero_u32(unsigned int* __restrict__ p, int n)
{
    int i = blockIdx.x * blockDim.x + threadIdx.x;
    if (i < n) p[i] = 0u;
}

// --- weight / layout prep ---------------------------------------------------
__global__ void pack_conv_w(const float* __restrict__ w, bf16* __restrict__ wpk,
                            int Cout, int Cin)
{
    int idx = blockIdx.x * blockDim.x + threadIdx.x;
    if (idx >= Cout * Cin * 9) return;
    int n = idx / (Cin * 9), rem = idx % (Cin * 9);
    int c = rem / 9, tap = rem % 9;
    wpk[(size_t)n * 9 * Cin + tap * Cin + c] = (bf16)w[idx];
}

__global__ void pack_deconv_w(const float* __restrict__ w, bf16* __restrict__ wd,
                              int Cin, int Cout, int s)
{
    int idx = blockIdx.x * blockDim.x + threadIdx.x;
    if (idx >= Cin * Cout * s * s) return;
    int c = idx / (Cout * s * s), rem = idx % (Cout * s * s);
    int o = rem / (s * s), ij = rem % (s * s);
    wd[(size_t)ij * Cin * Cout + c * Cout + o] = (bf16)w[idx];
}

__global__ void nchw_to_nhwc_bf16(const float* __restrict__ x, bf16* __restrict__ y,
                                  int C, int H, int W)
{
    int idx = blockIdx.x * blockDim.x + threadIdx.x;
    if (idx >= C * H * W) return;
    int c = idx / (H * W), p = idx % (H * W);
    int yy = p / W, xx = p % W;
    y[((size_t)(yy + 1) * (W + 2) + xx + 1) * C + c] = (bf16)x[idx];
}

// --- deconv (kernel==stride) + BN + ReLU, padded NHWC -----------------------
__global__ void deconv_bnrelu(const bf16* __restrict__ in, const bf16* __restrict__ wd,
                              bf16* __restrict__ out, int Hi, int Wi, int Cin, int Cout,
                              int s, float scale)
{
    int idx = blockIdx.x * blockDim.x + threadIdx.x;
    int Wo = Wi * s;
    if (idx >= Hi * s * Wo * Cout) return;
    int oc = idx % Cout, p = idx / Cout;
    int ox = p % Wo, oy = p / Wo;
    int iy = oy / s, ix = ox / s, i = oy % s, j = ox % s;
    const bf16* ib = in + ((size_t)(iy + 1) * (Wi + 2) + ix + 1) * Cin;
    const bf16* wb = wd + ((size_t)(i * s + j) * Cin) * Cout + oc;
    float a = 0.0f;
    for (int c = 0; c < Cin; ++c) a += (float)ib[c] * (float)wb[(size_t)c * Cout];
    out[((size_t)(oy + 1) * (Wo + 2) + ox + 1) * Cout + oc] = (bf16)fmaxf(a * scale, 0.0f);
}

// --- CBAM (x buffers are halo-padded; stats buffers dense) ------------------
__global__ void pool_avgmax(const bf16* __restrict__ x, float* __restrict__ avg,
                            float* __restrict__ mx, int H, int W, int C)
{
    int c = blockIdx.x, t = threadIdx.x;
    int HW = H * W, Wp = W + 2;
    float s = 0.0f, m = -1e30f;
    for (int p = t; p < HW; p += 256) {
        int yy = p / W, xx = p % W;
        float v = (float)x[((size_t)(yy + 1) * Wp + xx + 1) * C + c];
        s += v; m = fmaxf(m, v);
    }
    __shared__ float ss[256], sm[256];
    ss[t] = s; sm[t] = m; __syncthreads();
    for (int k = 128; k > 0; k >>= 1) {
        if (t < k) { ss[t] += ss[t + k]; sm[t] = fmaxf(sm[t], sm[t + k]); }
        __syncthreads();
    }
    if (t == 0) { avg[c] = ss[0] / (float)HW; mx[c] = sm[0]; }
}

__global__ void ca_mlp(const float* __restrict__ avg, const float* __restrict__ mx,
                       const float* __restrict__ w1, const float* __restrict__ w2,
                       float* __restrict__ ca, int C, int Ch)
{
    __shared__ float ha[32], hm[32];
    int t = threadIdx.x;
    if (t < Ch) {
        float sa = 0.0f, sm = 0.0f;
        for (int c = 0; c < C; ++c) { float w = w1[(size_t)t * C + c]; sa += avg[c] * w; sm += mx[c] * w; }
        ha[t] = fmaxf(sa, 0.0f); hm[t] = fmaxf(sm, 0.0f);
    }
    __syncthreads();
    if (t < C) {
        float v = 0.0f;
        for (int h = 0; h < Ch; ++h) v += (ha[h] + hm[h]) * w2[(size_t)t * Ch + h];
        ca[t] = 1.0f / (1.0f + expf(-v));
    }
}

__global__ void ca_apply_stats(bf16* __restrict__ x, const float* __restrict__ ca,
                               float* __restrict__ st, int H, int W, int C)
{
    int p = blockIdx.x * blockDim.x + threadIdx.x;
    if (p >= H * W) return;
    int yy = p / W, xx = p % W;
    bf16* row = x + ((size_t)(yy + 1) * (W + 2) + xx + 1) * C;
    float s = 0.0f, m = -1e30f;
    for (int c = 0; c < C; ++c) {
        float v = (float)row[c] * ca[c];
        row[c] = (bf16)v;
        s += v; m = fmaxf(m, v);
    }
    st[p * 2] = s / (float)C;
    st[p * 2 + 1] = m;
}

__global__ void sa_conv7(const float* __restrict__ st, const float* __restrict__ w,
                         float* __restrict__ sa, int H, int W)
{
    int p = blockIdx.x * blockDim.x + threadIdx.x;
    if (p >= H * W) return;
    int y = p / W, x = p % W;
    float a = 0.0f;
    for (int ch = 0; ch < 2; ++ch)
        for (int ky = 0; ky < 7; ++ky) {
            int iy = y + ky - 3;
            if (iy < 0 || iy >= H) continue;
            for (int kx = 0; kx < 7; ++kx) {
                int ix = x + kx - 3;
                if (ix < 0 || ix >= W) continue;
                a += st[(iy * W + ix) * 2 + ch] * w[ch * 49 + ky * 7 + kx];
            }
        }
    sa[p] = 1.0f / (1.0f + expf(-a));
}

__global__ void sa_apply(bf16* __restrict__ x, const float* __restrict__ sa,
                         int H, int W, int C)
{
    size_t idx = (size_t)blockIdx.x * blockDim.x + threadIdx.x;
    if (idx >= (size_t)H * W * C) return;
    int p = (int)(idx / C), c = (int)(idx % C);
    int yy = p / W, xx = p % W;
    size_t q = ((size_t)(yy + 1) * (W + 2) + xx + 1) * C + c;
    x[q] = (bf16)((float)x[q] * sa[p]);
}

// --- concat along C (both sources + dest halo-padded) -----------------------
__global__ void concat_c(const bf16* __restrict__ a, const bf16* __restrict__ b,
                         bf16* __restrict__ out, int H, int W, int C)
{
    size_t idx = (size_t)blockIdx.x * blockDim.x + threadIdx.x;
    if (idx >= (size_t)H * W * 2 * C) return;
    int c = (int)(idx % (2 * C));
    int p = (int)(idx / (2 * C));
    int yy = p / W, xx = p % W;
    size_t row = (size_t)(yy + 1) * (W + 2) + xx + 1;
    out[row * 2 * C + c] = (c < C) ? a[row * C + c] : b[row * C + (c - C)];
}

// --- heatmap tail (Cout=3), padded bf16 in, dense f32 CHW out ---------------
__global__ void conv3x3_small(const bf16* __restrict__ in, const float* __restrict__ w,
                              const float* __restrict__ bias, float* __restrict__ out,
                              int H, int W, int Cin, int Cout)
{
    int idx = blockIdx.x * blockDim.x + threadIdx.x;
    if (idx >= Cout * H * W) return;
    int co = idx / (H * W), p = idx % (H * W);
    int y = p / W, x = p % W;
    float a = bias[co];
    for (int tap = 0; tap < 9; ++tap) {
        int dy = tap / 3, dx = tap % 3;  // padded-space
        const bf16* ib = in + ((size_t)(y + dy) * (W + 2) + (x + dx)) * Cin;
        const float* wb = w + ((size_t)co * Cin) * 9 + tap;
        for (int c = 0; c < Cin; ++c) a += (float)ib[c] * wb[(size_t)c * 9];
    }
    out[idx] = a;
}

__global__ void score_from_hm(const float* __restrict__ hm, float* __restrict__ scores, int HW)
{
    int p = blockIdx.x * blockDim.x + threadIdx.x;
    if (p >= HW) return;
    float m = fmaxf(hm[p], fmaxf(hm[p + HW], hm[p + 2 * HW]));
    scores[p] = 1.0f / (1.0f + expf(-m));
}

__global__ __launch_bounds__(1024)
void topk500(float* __restrict__ scores, int* __restrict__ order, int HW)
{
    __shared__ float bv[1024];
    __shared__ int   bi[1024];
    int t = threadIdx.x;
    for (int it = 0; it < 500; ++it) {
        float best = -1e30f; int bidx = 0;
        for (int i = t; i < HW; i += 1024) {
            float v = scores[i];
            if (v > best) { best = v; bidx = i; }
        }
        bv[t] = best; bi[t] = bidx; __syncthreads();
        for (int s = 512; s > 0; s >>= 1) {
            if (t < s) {
                if (bv[t + s] > bv[t] || (bv[t + s] == bv[t] && bi[t + s] < bi[t])) {
                    bv[t] = bv[t + s]; bi[t] = bi[t + s];
                }
            }
            __syncthreads();
        }
        if (t == 0) { order[it] = bi[0]; scores[bi[0]] = -1e30f; }
        __syncthreads();
    }
}

// --- multiscale gather (sources halo-padded) --------------------------------
__global__ void gather_ms(const int* __restrict__ order, const bf16* __restrict__ xu,
                          const bf16* __restrict__ x0, const bf16* __restrict__ xd,
                          float* __restrict__ feats, float* __restrict__ pos, int W2, int C)
{
    int n = blockIdx.x, t = threadIdx.x;
    int id = order[n];
    int xc = id % W2, yc = id / W2;
    for (int j = 0; j < 27; ++j) {
        int i = j / 9, o = j % 9;
        int Ws = W2 >> i;
        int cx = (xc >> i) + (o / 3 - 1);
        int cy = (yc >> i) + (o % 3 - 1);
        cx = min(max(cx, 0), Ws - 1);
        cy = min(max(cy, 0), Ws - 1);
        const bf16* base = (i == 0 ? xu : (i == 1 ? x0 : xd));
        const bf16* f = base + ((size_t)(cy + 1) * (Ws + 2) + cx + 1) * C;
        float* dst = feats + ((size_t)n * 27 + j) * C;
        for (int c = t; c < C; c += 256) dst[c] = (float)f[c];
        if (t == 0) {
            pos[((size_t)n * 27 + j) * 2]     = (float)(cx << i);
            pos[((size_t)n * 27 + j) * 2 + 1] = (float)(cy << i);
        }
    }
}

// ---------------------------------------------------------------------------
static inline int cdiv(int a, int b) { return (a + b - 1) / b; }

extern "C" void kernel_launch(void* const* d_in, const int* in_sizes, int n_in,
                              void* d_out, int out_size, void* d_ws, size_t ws_size,
                              hipStream_t stream)
{
    // jax tree_flatten (sorted dict keys): params first, then x.
    // block: ca_w1, ca_w2, conv0, convs[...], sa_w
    const float* b0_caw1 = (const float*)d_in[0];
    const float* b0_caw2 = (const float*)d_in[1];
    const float* b0_c0   = (const float*)d_in[2];
    const float* b0_cv[5]= {(const float*)d_in[3], (const float*)d_in[4], (const float*)d_in[5],
                            (const float*)d_in[6], (const float*)d_in[7]};
    const float* b0_saw  = (const float*)d_in[8];
    const float* b1_caw1 = (const float*)d_in[9];
    const float* b1_caw2 = (const float*)d_in[10];
    const float* b1_c0   = (const float*)d_in[11];
    const float* b1_cv[5]= {(const float*)d_in[12], (const float*)d_in[13], (const float*)d_in[14],
                            (const float*)d_in[15], (const float*)d_in[16]};
    const float* b1_saw  = (const float*)d_in[17];
    const float* b2_caw1 = (const float*)d_in[18];
    const float* b2_caw2 = (const float*)d_in[19];
    const float* b2_c0   = (const float*)d_in[20];  // deconv w [256,128,4,4]
    const float* b2_cv0  = (const float*)d_in[21];
    const float* b2_saw  = (const float*)d_in[22];
    const float* hm_b1   = (const float*)d_in[23];
    const float* hm_b2   = (const float*)d_in[24];
    const float* hm_w1   = (const float*)d_in[25];
    const float* hm_w2   = (const float*)d_in[26];
    const float* up_w    = (const float*)d_in[27];  // [256,128,2,2]
    const float* x_in    = (const float*)d_in[28];  // [1,256,128,128]

    // ---- workspace bump allocator (halo-padded activations) ----
    char* ws = (char*)d_ws;
    size_t off = 0;
    auto alloc = [&](size_t bytes) -> void* {
        void* p = ws + off;
        off += (bytes + 255) & ~(size_t)255;
        return p;
    };
    const size_t N128 = (size_t)130 * 130 * 256;   // 128x128, C=256, padded
    const size_t N64  = (size_t)66 * 66 * 256;     // 64x64,  C=256, padded
    const size_t N256a= (size_t)258 * 258 * 128;   // 256x256, C=128, padded
    const size_t N256u= (size_t)258 * 258 * 256;   // 256x256, C=256, padded
    const size_t N256h= (size_t)258 * 258 * 64;    // 256x256, C=64,  padded

    bf16*  xa    = (bf16*)alloc(N128 * 2);
    bf16*  xb    = (bf16*)alloc(N128 * 2);
    bf16*  xda   = (bf16*)alloc(N64 * 2);
    bf16*  xdb   = (bf16*)alloc(N64 * 2);
    bf16*  t256a = (bf16*)alloc(N256a * 2);
    bf16*  t256b = (bf16*)alloc(N256a * 2);
    bf16*  xu    = (bf16*)alloc(N256u * 2);
    bf16*  hbuf  = (bf16*)alloc(N256h * 2);
    bf16*  wbuf  = (bf16*)alloc((size_t)256 * 2304 * 2);
    float* avgb  = (float*)alloc(256 * 4);
    float* mxb   = (float*)alloc(256 * 4);
    float* cab   = (float*)alloc(256 * 4);
    float* stb   = (float*)alloc((size_t)65536 * 2 * 4);
    float* sab   = (float*)alloc((size_t)65536 * 4);
    float* scr   = (float*)alloc((size_t)65536 * 4);
    int*   ordb  = (int*)alloc(500 * 4);

    auto zero = [&](void* p, size_t elems_bf16) {
        int n = (int)(elems_bf16 / 2);
        zero_u32<<<cdiv(n, 256), 256, 0, stream>>>((unsigned int*)p, n);
    };
    // zero all padded activation buffers so halos are 0 for the whole call
    zero(xa, N128); zero(xb, N128); zero(xda, N64); zero(xdb, N64);
    zero(t256a, N256a); zero(t256b, N256a); zero(xu, N256u); zero(hbuf, N256h);

    auto conv = [&](const bf16* in, int Win, const float* w, int Cin, int Cout,
                    int stride, const float* bias, bf16* out, int Ho, int Wo,
                    float scale, int relu) {
        pack_conv_w<<<cdiv(Cout * Cin * 9, 256), 256, 0, stream>>>(w, wbuf, Cout, Cin);
        if (Cout % 128 == 0) {
            dim3 g(Wo / 64, Ho, Cout / 128);
            conv3x3_wmma<4><<<g, 128, 0, stream>>>(in, wbuf, bias, out, Win, Wo,
                                                   Cin, Cout, stride, scale, relu);
        } else {
            dim3 g(Wo / 64, Ho, Cout / 64);
            conv3x3_wmma<2><<<g, 128, 0, stream>>>(in, wbuf, bias, out, Win, Wo,
                                                   Cin, Cout, stride, scale, relu);
        }
    };
    auto deconv = [&](const bf16* in, int Hi, int Wi, int Cin, int Cout, int s,
                      const float* w, bf16* out) {
        pack_deconv_w<<<cdiv(Cin * Cout * s * s, 256), 256, 0, stream>>>(w, wbuf, Cin, Cout, s);
        int tot = Hi * s * Wi * s * Cout;
        deconv_bnrelu<<<cdiv(tot, 256), 256, 0, stream>>>(in, wbuf, out, Hi, Wi, Cin, Cout, s, BNS);
    };
    auto cbam = [&](bf16* x, int H, int W, int C, const float* w1, const float* w2,
                    const float* sw) {
        pool_avgmax<<<C, 256, 0, stream>>>(x, avgb, mxb, H, W, C);
        ca_mlp<<<1, 256, 0, stream>>>(avgb, mxb, w1, w2, cab, C, C / 16);
        ca_apply_stats<<<cdiv(H * W, 256), 256, 0, stream>>>(x, cab, stb, H, W, C);
        sa_conv7<<<cdiv(H * W, 256), 256, 0, stream>>>(stb, sw, sab, H, W);
        sa_apply<<<cdiv(H * W * C, 256), 256, 0, stream>>>(x, sab, H, W, C);
    };

    // ---- forward ----
    nchw_to_nhwc_bf16<<<cdiv(256 * 128 * 128, 256), 256, 0, stream>>>(x_in, xa, 256, 128, 128);

    // b0: 6 convs 256->256 @128x128 + CBAM (ends in xa)
    conv(xa, 128, b0_c0, 256, 256, 1, nullptr, xb, 128, 128, BNS, 1);
    bf16 *pp = xb, *qq = xa;
    for (int i = 0; i < 5; ++i) {
        conv(pp, 128, b0_cv[i], 256, 256, 1, nullptr, qq, 128, 128, BNS, 1);
        bf16* t = pp; pp = qq; qq = t;
    }
    cbam(xa, 128, 128, 256, b0_caw1, b0_caw2, b0_saw);  // result in xa

    // b1: stride-2 then 5 convs @64x64 + CBAM (ends in xdb)
    conv(xa, 128, b1_c0, 256, 256, 2, nullptr, xda, 64, 64, BNS, 1);
    pp = xda; qq = xdb;
    for (int i = 0; i < 5; ++i) {
        conv(pp, 64, b1_cv[i], 256, 256, 1, nullptr, qq, 64, 64, BNS, 1);
        bf16* t = pp; pp = qq; qq = t;
    }
    cbam(xdb, 64, 64, 256, b1_caw1, b1_caw2, b1_saw);   // result in xdb

    // b2: deconv4 256->128 (64->256), conv 128->128, CBAM(C=128) -> t256b
    deconv(xdb, 64, 64, 256, 128, 4, b2_c0, t256a);
    conv(t256a, 256, b2_cv0, 128, 128, 1, nullptr, t256b, 256, 256, BNS, 1);
    cbam(t256b, 256, 256, 128, b2_caw1, b2_caw2, b2_saw);

    // up: deconv2 256->128 from x0 (xa) -> t256a; concat [b2 | up] -> xu
    deconv(xa, 128, 128, 256, 128, 2, up_w, t256a);
    concat_c<<<cdiv(256 * 256 * 256, 256), 256, 0, stream>>>(t256b, t256a, xu, 256, 256, 128);

    // heatmap head
    float* feats_o = (float*)d_out;
    float* pos_o   = (float*)d_out + (size_t)500 * 27 * 256;
    float* hm_out  = pos_o + (size_t)500 * 27 * 2;
    conv(xu, 256, hm_w1, 256, 64, 1, hm_b1, hbuf, 256, 256, BNS, 1);
    conv3x3_small<<<cdiv(3 * 256 * 256, 256), 256, 0, stream>>>(hbuf, hm_w2, hm_b2, hm_out,
                                                                256, 256, 64, 3);

    // scores + top-500 + gather
    score_from_hm<<<cdiv(65536, 256), 256, 0, stream>>>(hm_out, scr, 65536);
    topk500<<<1, 1024, 0, stream>>>(scr, ordb, 65536);
    gather_ms<<<500, 256, 0, stream>>>(ordb, xu, xa, xdb, feats_o, pos_o, 256, 256);
}
